// GATNet_24129126269181
// MI455X (gfx1250) — matrix operand order repaired and checked by reading, not compiled, not run on hardware
//
#include <hip/hip_runtime.h>
#include <hip/hip_bf16.h>

typedef __attribute__((ext_vector_type(16))) __bf16 bf16x16;
typedef __attribute__((ext_vector_type(8)))  __bf16 bf16x8;
typedef __attribute__((ext_vector_type(8)))  float  v8f;

#define DENSE   100
#define HEADS   10
#define HID     10
#define KPAD    128
#define NTILES  7      // 7*16 = 112 >= 100 columns
#define NGRAPH  256
#define OUTD    10
#define NEG_SLOPE 0.2f

static __device__ __forceinline__ unsigned short f32_to_bf16_bits(float f) {
  unsigned u = __float_as_uint(f);
  unsigned r = (u + 0x7FFFu + ((u >> 16) & 1u)) >> 16;  // round-to-nearest-even
  return (unsigned short)r;
}

static __device__ __forceinline__ void atomicMaxF(float* addr, float v) {
  // works for mixed-sign floats with -inf init
  if (v >= 0.0f) atomicMax((int*)addr, __float_as_int(v));
  else           atomicMin((unsigned int*)addr, __float_as_uint(v));
}

// ---------------- edge list with self loops ----------------
__global__ void k_build_edges(const int* __restrict__ ei, int E, int N,
                              int* __restrict__ srcd) {
  int t = blockIdx.x * blockDim.x + threadIdx.x;
  int Ep = E + N;
  if (t >= Ep) return;
  int s, d;
  if (t < E) { s = ei[t]; d = ei[E + t]; }
  else       { s = t - E; d = t - E; }
  srcd[t]      = s;
  srcd[Ep + t] = d;
}

// ---------------- pack activations f32 -> bf16, K padded to 128 ----------------
__global__ void k_pack_act(const float* __restrict__ in, int N, int Din,
                           unsigned short* __restrict__ out) {
  int t = blockIdx.x * blockDim.x + threadIdx.x;
  if (t >= N * KPAD) return;
  int k = t & (KPAD - 1);
  int n = t >> 7;
  float v = (k < Din) ? in[(size_t)n * Din + k] : 0.0f;
  out[t] = f32_to_bf16_bits(v);
}

// ---------------- pack W into per-lane WMMA B-operand vectors ----------------
// Layout: Bp[((ntile*4 + kstep)*32 + lane)*16 + j]
// B 32x16 bf16: col = ntile*16 + lane%16; K = kstep*32 + (j&7) + (lane>=16 ? 8 : 0) + (j>=8 ? 16 : 0)
__global__ void k_pack_W(const float* __restrict__ W, int Din,
                         unsigned short* __restrict__ Bp) {
  int t = blockIdx.x * blockDim.x + threadIdx.x;
  if (t >= NTILES * 4 * 32 * 16) return;
  int j     = t & 15;
  int lane  = (t >> 4) & 31;
  int kstep = (t >> 9) & 3;
  int ntile = t >> 11;
  int col = ntile * 16 + (lane & 15);
  int k   = kstep * 32 + (j & 7) + ((lane >= 16) ? 8 : 0) + ((j >= 8) ? 16 : 0);
  float v = (k < Din && col < DENSE) ? W[(size_t)k * DENSE + col] : 0.0f;
  Bp[t] = f32_to_bf16_bits(v);
}

// ---------------- GEMM: hfeat[N x 100] = Abf[N x 128] @ W, via WMMA bf16 ----------------
// One block = one 16-row tile of A, staged once into LDS via async copy and
// shared by 7 compute waves (one 16-col tile each).
__global__ void k_gemm(const unsigned short* __restrict__ Abf,
                       const unsigned short* __restrict__ Bp,
                       float* __restrict__ hfeat, int Nrows) {
  __shared__ unsigned short Atile[16 * KPAD];   // 4 KB
  int wave = threadIdx.x >> 5;   // 8 waves/block
  int lane = threadIdx.x & 31;
  int m0 = blockIdx.x * 16;
  if (m0 + 16 > Nrows) return;   // uniform over block

  // Cooperative async copy: 256 threads x 16 B = 4 KB (whole A tile).
  {
    unsigned byteoff = (unsigned)threadIdx.x * 16u;
    unsigned lds_off = (unsigned)(uintptr_t)(&Atile[0]) + byteoff;  // low 32b = LDS addr
    unsigned long long ga =
        (unsigned long long)(uintptr_t)(Abf + (size_t)m0 * KPAD) + byteoff;
    asm volatile("global_load_async_to_lds_b128 %0, %1, off"
                 :: "v"(lds_off), "v"(ga)
                 : "memory");
    asm volatile("s_wait_asynccnt 0x0" ::: "memory");
  }
  __syncthreads();
  if (wave >= NTILES) return;

  int koffA = (lane < 16) ? 0 : 8;   // A 16-bit layout: lanes 16-31 shifted by 8 in K
  const unsigned short* arow = &Atile[(lane & 15) * KPAD + koffA];

  v8f c = {};
#pragma unroll
  for (int ks = 0; ks < 4; ++ks) {
    bf16x8 alo = *(const bf16x8*)(arow + ks * 32);        // K = base+0..7   (+off)
    bf16x8 ahi = *(const bf16x8*)(arow + ks * 32 + 16);   // K = base+16..23 (+off)
    bf16x16 a = __builtin_shufflevector(alo, ahi,
        0, 1, 2, 3, 4, 5, 6, 7, 8, 9, 10, 11, 12, 13, 14, 15);
    bf16x16 b = *(const bf16x16*)(Bp + (((wave * 4 + ks) * 32 + lane) << 4));
    c = __builtin_amdgcn_wmma_f32_16x16x32_bf16(
        /*neg_a=*/false, a, /*neg_b=*/false, b,
        /*c_mod=*/(short)0, c, /*reuse_a=*/false, /*reuse_b=*/false);
  }

  int col   = wave * 16 + (lane & 15);
  int rbase = m0 + ((lane < 16) ? 0 : 8);   // C/D layout: VGPR r -> M=r / M=8+r
  if (col < DENSE) {
#pragma unroll
    for (int r = 0; r < 8; ++r)
      hfeat[(size_t)(rbase + r) * DENSE + col] = c[r];
  }
}

// ---------------- per-(node,head) attention scores ----------------
__global__ void k_scores(const float* __restrict__ h,
                         const float* __restrict__ a_src,
                         const float* __restrict__ a_dst,
                         float* __restrict__ ssrc, float* __restrict__ sdst, int N) {
  int t = blockIdx.x * blockDim.x + threadIdx.x;
  if (t >= N * HEADS) return;
  int hd = t % HEADS;
  int n  = t / HEADS;
  const float* hp = h + (size_t)n * DENSE + hd * HID;
  float s1 = 0.f, s2 = 0.f;
#pragma unroll
  for (int f = 0; f < HID; ++f) {
    float v = hp[f];
    s1 += v * a_src[hd * HID + f];
    s2 += v * a_dst[hd * HID + f];
  }
  ssrc[t] = s1;
  sdst[t] = s2;
}

// ---------------- init segment-softmax state + aggregation buffer ----------------
__global__ void k_init(float* __restrict__ amax, float* __restrict__ denom,
                       float* __restrict__ agg, int N) {
  int t = blockIdx.x * blockDim.x + threadIdx.x;
  if (t < N * DENSE) agg[t] = 0.0f;
  if (t < N * HEADS) { amax[t] = -__builtin_inff(); denom[t] = 0.0f; }
}

// ---------------- edge: alpha = leakyrelu(ssrc[s]+sdst[d]); segment max ----------------
__global__ void k_alpha(const int* __restrict__ srcd, int Ep,
                        const float* __restrict__ ssrc, const float* __restrict__ sdst,
                        float* __restrict__ alpha, float* __restrict__ amax) {
  int t = blockIdx.x * blockDim.x + threadIdx.x;
  if (t >= Ep * HEADS) return;
  int hd = t % HEADS;
  int e  = t / HEADS;
  int s = srcd[e];
  int d = srcd[Ep + e];
  float a = ssrc[s * HEADS + hd] + sdst[d * HEADS + hd];
  a = (a >= 0.0f) ? a : NEG_SLOPE * a;
  alpha[t] = a;
  atomicMaxF(&amax[d * HEADS + hd], a);
}

// ---------------- edge: ex = exp(alpha - amax[d]); segment sum (in-place ex) ----------------
__global__ void k_expsum(const int* __restrict__ srcd, int Ep,
                         float* __restrict__ alpha_ex,
                         const float* __restrict__ amax, float* __restrict__ denom) {
  int t = blockIdx.x * blockDim.x + threadIdx.x;
  if (t >= Ep * HEADS) return;
  int hd = t % HEADS;
  int e  = t / HEADS;
  int d = srcd[Ep + e];
  float ex = __expf(alpha_ex[t] - amax[d * HEADS + hd]);
  alpha_ex[t] = ex;
  atomicAdd(&denom[d * HEADS + hd], ex);
}

// ---------------- edge: agg[d] += h[s] * coef  (one thread per edge-feature) ----------------
__global__ void k_scatter(const int* __restrict__ srcd, int Ep,
                          const float* __restrict__ ex, const float* __restrict__ denom,
                          const float* __restrict__ h, float* __restrict__ agg) {
  int t = blockIdx.x * blockDim.x + threadIdx.x;
  int c = t & 127;
  int e = t >> 7;
  if (e >= Ep || c >= DENSE) return;
  int s = srcd[e];
  int d = srcd[Ep + e];
  int hd = c / HID;
  float coef = ex[e * HEADS + hd] / denom[d * HEADS + hd];
  atomicAdd(&agg[(size_t)d * DENSE + c], h[(size_t)s * DENSE + c] * coef);
}

// ---------------- bias + relu (in place) ----------------
__global__ void k_bias_relu(float* __restrict__ agg, const float* __restrict__ b, int N) {
  int t = blockIdx.x * blockDim.x + threadIdx.x;
  if (t >= N * DENSE) return;
  float v = agg[t] + b[t % DENSE];
  agg[t] = v > 0.0f ? v : 0.0f;
}

__global__ void k_zero(float* __restrict__ p, int n) {
  int t = blockIdx.x * blockDim.x + threadIdx.x;
  if (t < n) p[t] = 0.0f;
}

// ---------------- global mean pool (sum + count) ----------------
__global__ void k_pool(const float* __restrict__ h, const int* __restrict__ batch,
                       float* __restrict__ gsum, float* __restrict__ cnt, int N) {
  int t = blockIdx.x * blockDim.x + threadIdx.x;
  if (t >= N * DENSE) return;
  int c = t % DENSE;
  int n = t / DENSE;
  int g = batch[n];
  atomicAdd(&gsum[(size_t)g * DENSE + c], h[t]);
  if (c == 0) atomicAdd(&cnt[g], 1.0f);
}

// ---------------- final FC: logits = (gsum/cnt) @ W_fc + b_fc ----------------
__global__ void k_fc(const float* __restrict__ gsum, const float* __restrict__ cnt,
                     const float* __restrict__ Wfc, const float* __restrict__ bfc,
                     float* __restrict__ out) {
  int t = blockIdx.x * blockDim.x + threadIdx.x;
  if (t >= NGRAPH * OUTD) return;
  int o = t % OUTD;
  int g = t / OUTD;
  float inv = 1.0f / fmaxf(cnt[g], 1.0f);
  float acc = bfc[o];
  for (int c = 0; c < DENSE; ++c)
    acc += gsum[(size_t)g * DENSE + c] * inv * Wfc[c * OUTD + o];
  out[t] = acc;
}

static inline size_t align_up(size_t v, size_t a) { return (v + a - 1) & ~(a - 1); }
static inline int cdiv(int a, int b) { return (a + b - 1) / b; }

extern "C" void kernel_launch(void* const* d_in, const int* in_sizes, int n_in,
                              void* d_out, int out_size, void* d_ws, size_t ws_size,
                              hipStream_t stream) {
  const float* x     = (const float*)d_in[0];
  const int*   ei    = (const int*)d_in[1];
  const int*   batch = (const int*)d_in[2];
  const int N  = in_sizes[2];
  const int E  = in_sizes[1] / 2;
  const int Ep = E + N;

  // bump-allocate workspace
  char* w = (char*)d_ws;
  size_t off = 0;
  auto alloc = [&](size_t bytes) -> void* {
    void* p = w + off;
    off = align_up(off + bytes, 256);
    return p;
  };
  int*            srcd  = (int*)alloc((size_t)Ep * 2 * sizeof(int));
  unsigned short* hbf   = (unsigned short*)alloc((size_t)N * KPAD * 2);
  unsigned short* Bp    = (unsigned short*)alloc((size_t)NTILES * 4 * 32 * 16 * 2);
  float*          hfeat = (float*)alloc((size_t)N * DENSE * 4);
  float*          ssrc  = (float*)alloc((size_t)N * HEADS * 4);
  float*          sdst  = (float*)alloc((size_t)N * HEADS * 4);
  float*          amax  = (float*)alloc((size_t)N * HEADS * 4);
  float*          denom = (float*)alloc((size_t)N * HEADS * 4);
  float*          ex    = (float*)alloc((size_t)Ep * HEADS * 4);
  float*          agg   = (float*)alloc((size_t)N * DENSE * 4);
  float*          gsum  = (float*)alloc((size_t)NGRAPH * DENSE * 4);
  float*          cnt   = (float*)alloc((size_t)NGRAPH * 4);
  (void)ws_size; (void)n_in; (void)out_size;

  const int B = 256;
  k_build_edges<<<cdiv(Ep, B), B, 0, stream>>>(ei, E, N, srcd);

  const float* hsrc = x;
  for (int l = 0; l < 3; ++l) {
    const float* W   = (const float*)d_in[3 + 4 * l];
    const float* asr = (const float*)d_in[4 + 4 * l];
    const float* ads = (const float*)d_in[5 + 4 * l];
    const float* bb  = (const float*)d_in[6 + 4 * l];
    int Din = (l == 0) ? 128 : DENSE;

    k_pack_act<<<cdiv(N * KPAD, B), B, 0, stream>>>(hsrc, N, Din, hbf);
    k_pack_W<<<cdiv(NTILES * 4 * 32 * 16, B), B, 0, stream>>>(W, Din, Bp);
    k_gemm<<<cdiv(N, 16), B, 0, stream>>>(hbf, Bp, hfeat, N);
    k_scores<<<cdiv(N * HEADS, B), B, 0, stream>>>(hfeat, asr, ads, ssrc, sdst, N);
    k_init<<<cdiv(N * DENSE, B), B, 0, stream>>>(amax, denom, agg, N);
    k_alpha<<<cdiv(Ep * HEADS, B), B, 0, stream>>>(srcd, Ep, ssrc, sdst, ex, amax);
    k_expsum<<<cdiv(Ep * HEADS, B), B, 0, stream>>>(srcd, Ep, ex, amax, denom);
    k_scatter<<<cdiv(Ep * 128, B), B, 0, stream>>>(srcd, Ep, ex, denom, hfeat, agg);
    k_bias_relu<<<cdiv(N * DENSE, B), B, 0, stream>>>(agg, bb, N);
    hsrc = agg;
  }

  k_zero<<<cdiv(NGRAPH * DENSE, B), B, 0, stream>>>(gsum, NGRAPH * DENSE);
  k_zero<<<1, B, 0, stream>>>(cnt, NGRAPH);
  k_pool<<<cdiv(N * DENSE, B), B, 0, stream>>>(agg, batch, gsum, cnt, N);
  k_fc<<<cdiv(NGRAPH * OUTD, B), B, 0, stream>>>(
      gsum, cnt, (const float*)d_in[15], (const float*)d_in[16], (float*)d_out);
}
